// Attention_82892868813208
// MI455X (gfx1250) — matrix operand order repaired and checked by reading, not compiled, not run on hardware
//
#include <hip/hip_runtime.h>
#include <hip/hip_bf16.h>

// ---------------------------------------------------------------------------
// Attention layer for MI455X (gfx1250, wave32, WMMA 16x16x32 f16 -> f32 acc)
//   B=2, S=2048, HIDDEN=2048, HEADS=KV_HEADS=32, HD=64, causal.
// ---------------------------------------------------------------------------

typedef __attribute__((ext_vector_type(16))) _Float16 v16h;
typedef __attribute__((ext_vector_type(8)))  float    v8f;
typedef __attribute__((ext_vector_type(4)))  _Float16 v4h;
typedef __attribute__((ext_vector_type(4)))  int      v4i;

#define BATCH  2
#define SEQ    2048
#define HID    2048
#define NH     32
#define HD     64
#define QKVW   6144   // (32 + 2*32) * 64

// --- gfx1250 async global->LDS path (guarded: falls back to register copy) --
#if defined(__gfx1250__) && \
    __has_builtin(__builtin_amdgcn_global_load_async_to_lds_b128) && \
    __has_builtin(__builtin_amdgcn_s_wait_asynccnt)
#define USE_ASYNC_LDS 1
#else
#define USE_ASYNC_LDS 0
#endif

#if USE_ASYNC_LDS
#define ASG __attribute__((address_space(1)))
#define ASL __attribute__((address_space(3)))
template <int OFF>
__device__ __forceinline__ void async_b128(const _Float16* g, _Float16* l) {
  // INST_OFFSET applies to both the global and the LDS address (ISA 10, async).
  // Builtin signature (probe-confirmed via diagnostic): (global v4i*, local v4i*, imm, imm)
  __builtin_amdgcn_global_load_async_to_lds_b128(
      (ASG v4i*)const_cast<_Float16*>(g), (ASL v4i*)l, OFF, 0);
}
__device__ __forceinline__ void async_wait0() {
  __builtin_amdgcn_s_wait_asynccnt(0);
}
#endif

// --- CDNA5 WMMA fragment index helpers (per ISA 7.12.2 layouts) -------------
// A-matrix 16x32 f16: lane (half = lane/16, m = lane%16); VGPR v holds k pair:
//   k = (v>=4 ? 16 : 0) + half*8 + (v%4)*2, k+1
__device__ __forceinline__ int a_k(int v, int half) {
  return ((v & 4) << 2) + half * 8 + ((v & 3) << 1);
}
// B-matrix 32x16 f16: lane (half = lane/16, n = lane%16); VGPR v holds k pair:
//   k = (v>=4 ? 16 : 0) + half*8 + (v%4)*2, k+1
__device__ __forceinline__ int b_k(int v, int half) {
  return ((v & 4) << 2) + (half << 3) + ((v & 3) << 1);
}
// C/D 16x16 f32: lane holds rows m = v + 8*(lane/16), col n = lane%16.

// ---------------------------------------------------------------------------
// Kernel 1: fp32 -> fp16 conversion (4 elems/thread)
// ---------------------------------------------------------------------------
__global__ __launch_bounds__(256)
void f32_to_f16_kernel(const float* __restrict__ src, _Float16* __restrict__ dst) {
  size_t i = ((size_t)blockIdx.x * blockDim.x + threadIdx.x) * 4;
  float4 f = *(const float4*)(src + i);
  v4h h;
  h.x = (_Float16)f.x; h.y = (_Float16)f.y; h.z = (_Float16)f.z; h.w = (_Float16)f.w;
  *(v4h*)(dst + i) = h;
}

// ---------------------------------------------------------------------------
// Kernel 2/5: WMMA GEMM  C[M,N] = A[M,K] * B[K,N], f16 in, f32 acc.
//   Software-pipelined, double-buffered LDS; A tile moves HBM->LDS via the
//   async engine (ASYNCcnt) when available, overlapping the 2x2 WMMA block.
//   MODE 0: write f32 to Cout      MODE 1: scatter f16 into Q/K/V [b,h,s,d]
// Block: 256 threads = 8 waves (4x2), tile 128(M) x 64(N), K step 32.
// ---------------------------------------------------------------------------
template <int MODE>
__global__ __launch_bounds__(256)
void gemm_wmma_kernel(const _Float16* __restrict__ A, const _Float16* __restrict__ Bmat,
                      float* __restrict__ Cout,
                      _Float16* __restrict__ Qo, _Float16* __restrict__ Ko,
                      _Float16* __restrict__ Vo,
                      int M, int N, int K) {
  __shared__ _Float16 As[2][128 * 32];   // [m][k] row-major, double buffered
  __shared__ _Float16 Bs[2][64 * 32];    // [n][k] transposed, double buffered

  const int tid  = threadIdx.x;
  const int lane = tid & 31;
  const int wave = tid >> 5;
  const int wm   = wave & 3;             // 0..3 -> 32-row sub-block
  const int wn   = wave >> 2;            // 0..1 -> 32-col sub-block
  const int half = lane >> 4;
  const int l16  = lane & 15;

  const int m_base = blockIdx.y * 128;
  const int n_base = blockIdx.x * 64;

  // per-thread staging coordinates
  const int a_m  = tid >> 1;             // 0..127
  const int a_kp = (tid & 1) * 16;       // 0 / 16 halves
  const int b_kk = tid & 31;             // k row
  const int b_n  = (tid >> 5) * 8;       // 8 n-columns
  const _Float16* Abase = A + (size_t)(m_base + a_m) * K + a_kp;
  const _Float16* Bbase = Bmat + (size_t)b_kk * N + n_base + b_n;

  v8f acc[2][2] = {};
  const int T = K / 32;

  // ---- prologue: stage tile 0 into buffer 0 --------------------------------
#if USE_ASYNC_LDS
  async_b128<0>(Abase, &As[0][a_m * 32 + a_kp]);
  async_b128<16>(Abase, &As[0][a_m * 32 + a_kp]);
#else
  {
    uint4 a0 = *(const uint4*)(Abase);
    uint4 a1 = *(const uint4*)(Abase + 8);
    *(uint4*)&As[0][a_m * 32 + a_kp]     = a0;
    *(uint4*)&As[0][a_m * 32 + a_kp + 8] = a1;
  }
#endif
  {
    uint4 d = *(const uint4*)(Bbase);
    const _Float16* hp = (const _Float16*)&d;
#pragma unroll
    for (int i = 0; i < 8; ++i) Bs[0][(b_n + i) * 32 + b_kk] = hp[i];
  }
#if USE_ASYNC_LDS
  async_wait0();
#endif
  __syncthreads();

  for (int t = 0; t < T; ++t) {
    const int cur = t & 1;
    const int nxt = cur ^ 1;

    // ---- issue next tile (overlaps with this tile's WMMA) ------------------
#if !USE_ASYNC_LDS
    uint4 a0, a1;
#endif
    uint4 bnext;
    if (t + 1 < T) {
      const int k1 = (t + 1) * 32;
#if USE_ASYNC_LDS
      async_b128<0>(Abase + k1, &As[nxt][a_m * 32 + a_kp]);
      async_b128<16>(Abase + k1, &As[nxt][a_m * 32 + a_kp]);
#else
      a0 = *(const uint4*)(Abase + k1);
      a1 = *(const uint4*)(Abase + k1 + 8);
#endif
      bnext = *(const uint4*)(Bbase + (size_t)k1 * N);
    }
    if (t + 2 < T)   // gfx1250 global_prefetch for the tile after next
      __builtin_prefetch(Abase + (t + 2) * 32, 0, 1);

    // ---- fragments & 2x2 WMMA from current buffer --------------------------
    v16h afrag[2], bfrag[2];
#pragma unroll
    for (int mf = 0; mf < 2; ++mf) {
      const unsigned* arow = (const unsigned*)(&As[cur][(wm * 32 + mf * 16 + l16) * 32]);
      unsigned* av = (unsigned*)&afrag[mf];
#pragma unroll
      for (int v = 0; v < 8; ++v) av[v] = arow[a_k(v, half) >> 1];
    }
#pragma unroll
    for (int nf = 0; nf < 2; ++nf) {
      const unsigned* brow = (const unsigned*)(&Bs[cur][(wn * 32 + nf * 16 + l16) * 32]);
      unsigned* bv = (unsigned*)&bfrag[nf];
#pragma unroll
      for (int v = 0; v < 8; ++v) bv[v] = brow[b_k(v, half) >> 1];
    }
#pragma unroll
    for (int mf = 0; mf < 2; ++mf)
#pragma unroll
      for (int nf = 0; nf < 2; ++nf)
        acc[mf][nf] = __builtin_amdgcn_wmma_f32_16x16x32_f16(
            false, afrag[mf], false, bfrag[nf], (short)0, acc[mf][nf], false, false);

    // ---- commit next tile into the other buffer ----------------------------
    if (t + 1 < T) {
#if !USE_ASYNC_LDS
      *(uint4*)&As[nxt][a_m * 32 + a_kp]     = a0;
      *(uint4*)&As[nxt][a_m * 32 + a_kp + 8] = a1;
#endif
      const _Float16* hp = (const _Float16*)&bnext;
#pragma unroll
      for (int i = 0; i < 8; ++i) Bs[nxt][(b_n + i) * 32 + b_kk] = hp[i];
#if USE_ASYNC_LDS
      async_wait0();
#endif
    }
    __syncthreads();
  }

  // --- epilogue --------------------------------------------------------------
#pragma unroll
  for (int mf = 0; mf < 2; ++mf) {
#pragma unroll
    for (int nf = 0; nf < 2; ++nf) {
#pragma unroll
      for (int v = 0; v < 8; ++v) {
        const int m = m_base + wm * 32 + mf * 16 + v + 8 * half;
        const int n = n_base + wn * 32 + nf * 16 + l16;
        const float val = acc[mf][nf][v];
        if (MODE == 0) {
          Cout[(size_t)m * N + n] = val;
        } else {
          const int sec = n >> 11;           // 0:Q 1:K 2:V
          const int hh  = (n & 2047) >> 6;   // head
          const int d   = n & 63;
          const int b   = m >> 11;           // row / SEQ
          const int s   = m & 2047;
          _Float16* buf = (sec == 0) ? Qo : (sec == 1) ? Ko : Vo;
          buf[(((size_t)b * NH + hh) * SEQ + s) * HD + d] = (_Float16)val;
        }
      }
    }
  }
}

// ---------------------------------------------------------------------------
// Kernel 3: RoPE in-place on Q and K ([b,h,s,d] f16), cos/sin f32 [S,HD]
// ---------------------------------------------------------------------------
__global__ __launch_bounds__(256)
void rope_kernel(_Float16* __restrict__ Q, _Float16* __restrict__ Kb,
                 const float* __restrict__ cosv, const float* __restrict__ sinv) {
  const size_t i = (size_t)blockIdx.x * blockDim.x + threadIdx.x; // B*NH*SEQ*32
  const int d     = (int)(i & 31);
  const size_t r  = i >> 5;                 // (b*NH + h)*SEQ + s
  const int s     = (int)(r & (SEQ - 1));
  const size_t base = r * HD;

  const float c1 = cosv[s * HD + d],      s1 = sinv[s * HD + d];
  const float c2 = cosv[s * HD + d + 32], s2 = sinv[s * HD + d + 32];

  const float q1 = (float)Q[base + d], q2 = (float)Q[base + d + 32];
  Q[base + d]      = (_Float16)(q1 * c1 - q2 * s1);
  Q[base + d + 32] = (_Float16)(q2 * c2 + q1 * s2);

  const float k1 = (float)Kb[base + d], k2 = (float)Kb[base + d + 32];
  Kb[base + d]      = (_Float16)(k1 * c1 - k2 * s1);
  Kb[base + d + 32] = (_Float16)(k2 * c2 + k1 * s2);
}

// ---------------------------------------------------------------------------
// Kernel 4: causal flash attention.
//   grid = (SEQ/64, BATCH*NH), block = 128 (4 waves), wave = 16 q-rows.
//   Key tiles of 32; K-tile staged HBM->LDS via async engine; S = Q K^T
//   (2 WMMA / 16-key sub-tile), online softmax with half-wave shfl_xor row
//   reductions, O += P V (4 WMMA).
// ---------------------------------------------------------------------------
__global__ __launch_bounds__(128)
void flash_attn_kernel(const _Float16* __restrict__ Q, const _Float16* __restrict__ Kb,
                       const _Float16* __restrict__ Vb, _Float16* __restrict__ O) {
  __shared__ _Float16 Ks[32 * 64];     // [key][d]
  __shared__ _Float16 Vt[64 * 32];     // [d][key] (transposed)
  __shared__ _Float16 Ps[4 * 16 * 32]; // per-wave P staging [m][key]

  const int tid  = threadIdx.x;
  const int lane = tid & 31;
  const int wave = tid >> 5;
  const int half = lane >> 4;
  const int l16  = lane & 15;
  const int bh   = blockIdx.y;         // b*NH + h
  const int b    = bh >> 5;
  const int h    = bh & 31;
  const int q0   = blockIdx.x * 64 + wave * 16;
  const size_t head_base = (size_t)bh * SEQ * HD;

  // Q fragments (A layout), k-chunks d=[0,32) and [32,64)
  v16h qfrag[2];
  {
    const unsigned* qp = (const unsigned*)(Q + head_base + (size_t)(q0 + l16) * HD);
    unsigned* qa = (unsigned*)&qfrag[0];
    unsigned* qb = (unsigned*)&qfrag[1];
#pragma unroll
    for (int v = 0; v < 8; ++v) {
      qa[v] = qp[(a_k(v, half)) >> 1];
      qb[v] = qp[(32 + a_k(v, half)) >> 1];
    }
  }

  v8f o_acc[4] = {};
  float m_run[8], l_run[8];
#pragma unroll
  for (int v = 0; v < 8; ++v) { m_run[v] = -1e30f; l_run[v] = 0.0f; }

  const int nkt = (blockIdx.x + 1) * 2;      // key tiles covering the diagonal
  for (int kt = 0; kt < nkt; ++kt) {
    const int key_base = kt * 32;
    __syncthreads();
    // --- stage K tile [key][d]: async HBM->LDS when available --------------
    {
      const int row  = tid >> 2;            // 0..31
      const int part = (tid & 3) * 16;      // halves along d
      const _Float16* gsrc = Kb + head_base + (size_t)(key_base + row) * HD + part;
      _Float16* ldst = Ks + row * 64 + part;
#if USE_ASYNC_LDS
      async_b128<0>(gsrc, ldst);
      async_b128<16>(gsrc, ldst);
#else
      ((uint4*)ldst)[0] = ((const uint4*)gsrc)[0];
      ((uint4*)ldst)[1] = ((const uint4*)gsrc)[1];
#endif
    }
    // --- stage V tile transposed [d][key] ---------------------------------
    {
      const int key   = tid >> 2;           // 0..31
      const int dpart = (tid & 3) * 16;
      const uint4* src = (const uint4*)(Vb + head_base + (size_t)(key_base + key) * HD + dpart);
      uint4 d0 = src[0], d1 = src[1];
      const _Float16* h0 = (const _Float16*)&d0;
      const _Float16* h1 = (const _Float16*)&d1;
#pragma unroll
      for (int i = 0; i < 8; ++i) Vt[(dpart + i) * 32 + key] = h0[i];
#pragma unroll
      for (int i = 0; i < 8; ++i) Vt[(dpart + 8 + i) * 32 + key] = h1[i];
    }
#if USE_ASYNC_LDS
    async_wait0();
#endif
    __syncthreads();

    // --- S = Q K^T (per 16-key sub-tile) ----------------------------------
    v8f sfrag[2];
#pragma unroll
    for (int nt = 0; nt < 2; ++nt) {
      v16h kf0, kf1;
      const unsigned* kp = (const unsigned*)(Ks + (nt * 16 + l16) * 64);
      unsigned* k0v = (unsigned*)&kf0;
      unsigned* k1v = (unsigned*)&kf1;
#pragma unroll
      for (int v = 0; v < 8; ++v) {
        k0v[v] = kp[(b_k(v, half)) >> 1];
        k1v[v] = kp[(32 + b_k(v, half)) >> 1];
      }
      v8f s = {};
      s = __builtin_amdgcn_wmma_f32_16x16x32_f16(false, qfrag[0], false, kf0, (short)0, s, false, false);
      s = __builtin_amdgcn_wmma_f32_16x16x32_f16(false, qfrag[1], false, kf1, (short)0, s, false, false);
      sfrag[nt] = s;
    }

    // --- scale + causal mask ----------------------------------------------
#pragma unroll
    for (int nt = 0; nt < 2; ++nt)
#pragma unroll
      for (int v = 0; v < 8; ++v) {
        const int key  = key_base + nt * 16 + l16;
        const int qrow = q0 + v + 8 * half;
        const float sv = sfrag[nt][v] * 0.125f;   // 1/sqrt(64)
        sfrag[nt][v] = (key > qrow) ? -1e30f : sv;
      }

    // --- online softmax (rows live in 16-lane half-groups) ----------------
    float p0[8], p1[8];
#pragma unroll
    for (int v = 0; v < 8; ++v) {
      float t = fmaxf(sfrag[0][v], sfrag[1][v]);
      t = fmaxf(t, __shfl_xor(t, 1));
      t = fmaxf(t, __shfl_xor(t, 2));
      t = fmaxf(t, __shfl_xor(t, 4));
      t = fmaxf(t, __shfl_xor(t, 8));
      const float mnew = fmaxf(m_run[v], t);
      const float fac  = __expf(m_run[v] - mnew);
      p0[v] = __expf(sfrag[0][v] - mnew);
      p1[v] = __expf(sfrag[1][v] - mnew);
      float rs = p0[v] + p1[v];
      rs += __shfl_xor(rs, 1);
      rs += __shfl_xor(rs, 2);
      rs += __shfl_xor(rs, 4);
      rs += __shfl_xor(rs, 8);
      l_run[v] = l_run[v] * fac + rs;
      m_run[v] = mnew;
#pragma unroll
      for (int c = 0; c < 4; ++c) o_acc[c][v] *= fac;
    }

    // --- re-layout P (C layout -> row-major) through per-wave LDS ---------
    _Float16* myP = Ps + wave * (16 * 32);
#pragma unroll
    for (int v = 0; v < 8; ++v) {
      const int m = v + 8 * half;
      myP[m * 32 + l16]      = (_Float16)p0[v];
      myP[m * 32 + 16 + l16] = (_Float16)p1[v];
    }
    v16h pfrag;
    {
      const unsigned* pp = (const unsigned*)(myP + l16 * 32);
      unsigned* pv = (unsigned*)&pfrag;
#pragma unroll
      for (int v = 0; v < 8; ++v) pv[v] = pp[a_k(v, half) >> 1];
    }

    // --- O += P * V (4 d-chunks of 16) ------------------------------------
#pragma unroll
    for (int c = 0; c < 4; ++c) {
      v16h vf;
      const unsigned* vp = (const unsigned*)(Vt + (c * 16 + l16) * 32);
      unsigned* vv = (unsigned*)&vf;
#pragma unroll
      for (int v = 0; v < 8; ++v) vv[v] = vp[b_k(v, half) >> 1];
      o_acc[c] = __builtin_amdgcn_wmma_f32_16x16x32_f16(
          false, pfrag, false, vf, (short)0, o_acc[c], false, false);
    }
  }

  // --- normalize + store O as [b*S+s][NH*HD] f16 ---------------------------
#pragma unroll
  for (int v = 0; v < 8; ++v) {
    const float inv = 1.0f / l_run[v];
    const int s = q0 + v + 8 * half;
    const size_t rowoff = (size_t)(b * SEQ + s) * (NH * HD) + h * HD;
#pragma unroll
    for (int c = 0; c < 4; ++c)
      O[rowoff + c * 16 + l16] = (_Float16)(o_acc[c][v] * inv);
  }
}

// ---------------------------------------------------------------------------
// Host launcher
// ---------------------------------------------------------------------------
extern "C" void kernel_launch(void* const* d_in, const int* in_sizes, int n_in,
                              void* d_out, int out_size, void* d_ws, size_t ws_size,
                              hipStream_t stream) {
  (void)in_sizes; (void)n_in; (void)out_size; (void)ws_size;
  const float* hidden = (const float*)d_in[0];
  const float* cosv   = (const float*)d_in[1];
  const float* sinv   = (const float*)d_in[2];
  const float* wqkv   = (const float*)d_in[3];
  const float* wo     = (const float*)d_in[4];

  char* ws = (char*)d_ws;
  // Workspace layout (bytes):
  _Float16* hA   = (_Float16*)(ws);               // 8,388,608  f16 -> 16 MiB
  _Float16* wq16 = (_Float16*)(ws +  16777216);   // 12,582,912 f16 -> 24 MiB
  _Float16* wo16 = (_Float16*)(ws +  41943040);   //  4,194,304 f16 ->  8 MiB
  _Float16* Qb   = (_Float16*)(ws +  50331648);   //  8,388,608 f16 -> 16 MiB
  _Float16* Kb   = (_Float16*)(ws +  67108864);   //  8,388,608 f16 -> 16 MiB
  _Float16* Vb   = (_Float16*)(ws +  83886080);   //  8,388,608 f16 -> 16 MiB
  _Float16* Ob   = (_Float16*)(ws + 100663296);   //  8,388,608 f16 -> 16 MiB

  const int M = BATCH * SEQ;                      // 4096

  // 1) convert operands to f16
  f32_to_f16_kernel<<< (M * HID)    / 1024, 256, 0, stream >>>(hidden, hA);
  f32_to_f16_kernel<<< (HID * QKVW) / 1024, 256, 0, stream >>>(wqkv, wq16);
  f32_to_f16_kernel<<< (HID * HID)  / 1024, 256, 0, stream >>>(wo, wo16);

  // 2) QKV projection, scattering straight into per-head Q/K/V buffers
  gemm_wmma_kernel<1><<< dim3(QKVW / 64, M / 128), 256, 0, stream >>>(
      hA, wq16, nullptr, Qb, Kb, Vb, M, QKVW, HID);

  // 3) RoPE in place on Q, K
  rope_kernel<<< (BATCH * NH * SEQ * 32) / 256, 256, 0, stream >>>(Qb, Kb, cosv, sinv);

  // 4) causal flash attention -> Ob [B*S][NH*HD]
  flash_attn_kernel<<< dim3(SEQ / 64, BATCH * NH), 128, 0, stream >>>(Qb, Kb, Vb, Ob);

  // 5) output projection -> f32 d_out
  gemm_wmma_kernel<0><<< dim3(HID / 64, M / 128), 256, 0, stream >>>(
      Ob, wo16, (float*)d_out, nullptr, nullptr, nullptr, M, HID, HID);
}
